// MoELayer_13898514170500
// MI455X (gfx1250) — matrix operand order repaired and checked by reading, not compiled, not run on hardware
//
#include <hip/hip_runtime.h>
#include <hip/hip_bf16.h>
#include <cstdint>
#include <cstddef>

// ---------------- problem constants (from reference) ----------------
#define MOE_E 8
#define MOE_K 2
#define MOE_D 1024
#define MOE_H 4096

// ---------------- tiling ----------------
#define TM 128    // tokens per block tile
#define TN 128    // output cols per block tile
#define KS 32     // K per WMMA step (bf16 16x16x32)
#define KP 40     // LDS K pitch (bf16 elems): 80B rows -> conflict-free b128 reads

typedef __attribute__((ext_vector_type(16))) __bf16 v16bf;
typedef __attribute__((ext_vector_type(8)))  __bf16 v8bf;
typedef __attribute__((ext_vector_type(8)))  float  v8f;

typedef __attribute__((address_space(3))) char lds_char;

// LDS byte offset of a generic pointer into the block's LDS segment
__device__ __forceinline__ unsigned ldsoff(void* p) {
    return (unsigned)(unsigned long long)(lds_char*)p;
}

// async global->LDS 16B copy (per-lane addresses), tracked on ASYNCcnt
__device__ __forceinline__ void async_g2l_b128(unsigned lds_b, unsigned g_b, const void* gbase) {
    asm volatile("global_load_async_to_lds_b128 %0, %1, %2"
                 : : "v"(lds_b), "v"(g_b), "s"(gbase) : "memory");
}
__device__ __forceinline__ void wait_async0() {
    asm volatile("s_wait_asynccnt 0x0" : : : "memory");
}

// pack two f32 -> packed bf16x2 dword
__device__ __forceinline__ unsigned pack_bf16x2(float a, float b) {
    union { __bf16 h[2]; unsigned u; } pk;
    pk.h[0] = (__bf16)a;
    pk.h[1] = (__bf16)b;
    return pk.u;
}

// =====================================================================
// prep: zero output, convert x -> bf16, zero per-expert counters
// =====================================================================
__global__ __launch_bounds__(256) void moe_prep(
    const float* __restrict__ x, __bf16* __restrict__ xb,
    float* __restrict__ out, int* __restrict__ counts, size_t total)
{
    size_t i = (size_t)blockIdx.x * 256 + threadIdx.x;
    if (i < total) {
        xb[i]  = (__bf16)x[i];
        out[i] = 0.0f;
    }
    if (i < MOE_E) counts[i] = 0;
}

// =====================================================================
// router: one wave32 per token
// =====================================================================
__global__ __launch_bounds__(32) void moe_router(
    const float* __restrict__ x, const float* __restrict__ gw,
    const float* __restrict__ gb, float* __restrict__ comb,
    int* __restrict__ lists, int* __restrict__ counts, int N)
{
    const int n    = blockIdx.x;
    const int lane = threadIdx.x;

    float acc[MOE_E];
#pragma unroll
    for (int e = 0; e < MOE_E; ++e) acc[e] = 0.0f;

    const float* xr = x + (size_t)n * MOE_D;
    for (int i = 0; i < MOE_D / 32; ++i) {
        int d = i * 32 + lane;
        float xv = xr[d];
        const float* wr = gw + (size_t)d * MOE_E;
#pragma unroll
        for (int e = 0; e < MOE_E; ++e) acc[e] += xv * wr[e];
    }
#pragma unroll
    for (int e = 0; e < MOE_E; ++e) {
#pragma unroll
        for (int off = 16; off > 0; off >>= 1)
            acc[e] += __shfl_xor(acc[e], off, 32);
    }

    if (lane == 0) {
        float l[MOE_E];
#pragma unroll
        for (int e = 0; e < MOE_E; ++e) l[e] = acc[e] + gb[e];   // TEMP == 1
        float m = l[0];
#pragma unroll
        for (int e = 1; e < MOE_E; ++e) m = fmaxf(m, l[e]);
        float ex[MOE_E], Z = 0.0f;
#pragma unroll
        for (int e = 0; e < MOE_E; ++e) { ex[e] = expf(l[e] - m); Z += ex[e]; }

        int e1 = 0;
#pragma unroll
        for (int e = 1; e < MOE_E; ++e) if (ex[e] > ex[e1]) e1 = e;
        int e2 = (e1 == 0) ? 1 : 0;
#pragma unroll
        for (int e = 0; e < MOE_E; ++e) if (e != e1 && ex[e] > ex[e2]) e2 = e;

        float g1 = ex[e1] / Z, g2 = ex[e2] / Z;
        float norm = g1 + g2 + 1e-9f;
        float cw[MOE_E];
#pragma unroll
        for (int e = 0; e < MOE_E; ++e) cw[e] = 0.0f;
        cw[e1] = g1 / norm;
        cw[e2] = g2 / norm;
#pragma unroll
        for (int e = 0; e < MOE_E; ++e) comb[(size_t)n * MOE_E + e] = cw[e];

        int p1 = atomicAdd(&counts[e1], 1);
        lists[(size_t)e1 * N + p1] = n;
        int p2 = atomicAdd(&counts[e2], 1);
        lists[(size_t)e2 * N + p2] = n;
    }
}

// =====================================================================
// bases: exclusive prefix sum over 8 counts
// =====================================================================
__global__ void moe_bases(const int* __restrict__ counts, int* __restrict__ bases)
{
    if (threadIdx.x == 0 && blockIdx.x == 0) {
        int run = 0;
        for (int e = 0; e < MOE_E; ++e) { bases[e] = run; run += counts[e]; }
    }
}

// =====================================================================
// ffn1: h[slot, :] = gelu(x[token] @ w1[e] + b1[e]), bf16 output
// grid: (H/TN, ceil(N/TM), E), block 256 (8 waves; wave tile 32x64)
// =====================================================================
__global__ __launch_bounds__(256) void moe_ffn1(
    const __bf16* __restrict__ xb, const float* __restrict__ w1,
    const float* __restrict__ b1, const int* __restrict__ lists,
    const int* __restrict__ counts, const int* __restrict__ bases,
    __bf16* __restrict__ hbuf, int N)
{
    const int e     = blockIdx.z;
    const int cnt   = counts[e];
    const int slot0 = blockIdx.y * TM;
    if (slot0 >= cnt) return;
    const int nblk  = blockIdx.x * TN;
    const int base  = bases[e];

    __shared__ __bf16 ldsA[TM][KP];
    __shared__ __bf16 ldsB[TN][KP];
    __shared__ int    ldsTok[TM];

    const int t    = threadIdx.x;
    const int lane = t & 31;
    const int wid  = t >> 5;

    if (t < TM) {
        int s = slot0 + t;
        if (s >= cnt) s = cnt - 1;           // duplicate last valid token (stores guarded)
        ldsTok[t] = lists[(size_t)e * N + s];
    }
    __syncthreads();

    // A fill: 256 thr -> 128 rows x 2 chunks of 16 bf16 (2 async b128 each)
    const int arow  = t >> 1;
    const int akoff = (t & 1) * 16;
    const unsigned agb0 = (unsigned)(((size_t)ldsTok[arow] * MOE_D + akoff) * sizeof(__bf16));
    const unsigned alds0 = ldsoff(&ldsA[arow][akoff]);

    // B fill: 256 thr -> 16 n-groups(8) x 16 k-pairs(2)
    const int bn = (t & 15) * 8;
    const int bk = (t >> 4) * 2;
    const float* w1e = w1 + (size_t)e * MOE_D * MOE_H + (size_t)bk * MOE_H + nblk + bn;

    const int wm  = (wid & 3) * 32;   // 4 waves along M (32 rows each)
    const int wn  = (wid >> 2) * 64;  // 2 waves along N (64 cols each)
    const int l15 = lane & 15;
    const int lh  = lane >> 4;

    v8f acc[2][4] = {};

    for (int k0 = 0; k0 < MOE_D; k0 += KS) {
        // global weight loads (overlap previous compute)
        const float* wp = w1e + (size_t)k0 * MOE_H;
        float4 r0a = *(const float4*)(wp);
        float4 r0b = *(const float4*)(wp + 4);
        float4 r1a = *(const float4*)(wp + MOE_H);
        float4 r1b = *(const float4*)(wp + MOE_H + 4);

        __syncthreads();   // previous compute done; LDS free to overwrite

        // async A tile (bf16 copy, no VGPR round trip)
        async_g2l_b128(alds0,      agb0 + (unsigned)(k0 * 2),      xb);
        async_g2l_b128(alds0 + 16, agb0 + (unsigned)(k0 * 2) + 16, xb);

        // B tile: f32 -> bf16, transposed [n][k], packed dword stores
        float f0[8] = {r0a.x, r0a.y, r0a.z, r0a.w, r0b.x, r0b.y, r0b.z, r0b.w};
        float f1[8] = {r1a.x, r1a.y, r1a.z, r1a.w, r1b.x, r1b.y, r1b.z, r1b.w};
#pragma unroll
        for (int j = 0; j < 8; ++j)
            *(unsigned*)&ldsB[bn + j][bk] = pack_bf16x2(f0[j], f1[j]);

        wait_async0();
        __syncthreads();

        v16bf af[2], bfr[4];
#pragma unroll
        for (int ti = 0; ti < 2; ++ti) {
            int row = wm + ti * 16 + l15;
            v8bf lo = *(const v8bf*)&ldsA[row][lh * 8];
            v8bf hi = *(const v8bf*)&ldsA[row][lh * 8 + 16];
            af[ti] = __builtin_shufflevector(lo, hi, 0,1,2,3,4,5,6,7,8,9,10,11,12,13,14,15);
        }
#pragma unroll
        for (int tj = 0; tj < 4; ++tj) {
            int col = wn + tj * 16 + l15;
            v8bf lo = *(const v8bf*)&ldsB[col][lh * 8];
            v8bf hi = *(const v8bf*)&ldsB[col][lh * 8 + 16];
            bfr[tj] = __builtin_shufflevector(lo, hi, 0,1,2,3,4,5,6,7,8,9,10,11,12,13,14,15);
        }
#pragma unroll
        for (int ti = 0; ti < 2; ++ti)
#pragma unroll
            for (int tj = 0; tj < 4; ++tj)
                acc[ti][tj] = __builtin_amdgcn_wmma_f32_16x16x32_bf16(
                    false, af[ti], false, bfr[tj], (short)0, acc[ti][tj], false, false);
    }

    // epilogue: bias + exact GELU, store bf16
#pragma unroll
    for (int tj = 0; tj < 4; ++tj) {
        int colg = nblk + wn + tj * 16 + l15;
        float bb = b1[(size_t)e * MOE_H + colg];
#pragma unroll
        for (int ti = 0; ti < 2; ++ti) {
#pragma unroll
            for (int r = 0; r < 8; ++r) {
                int rl = wm + ti * 16 + lh * 8 + r;
                int s  = slot0 + rl;
                if (s < cnt) {
                    float v = acc[ti][tj][r] + bb;
                    v = 0.5f * v * (1.0f + erff(v * 0.70710678118654752f));
                    hbuf[(size_t)(base + s) * MOE_H + colg] = (__bf16)v;
                }
            }
        }
    }
}

// =====================================================================
// ffn2: out[token, :] += comb * (h[slot] @ w2[e] + b2[e])  (f32 atomics)
// grid: (D/TN, ceil(N/TM), E), block 256
// =====================================================================
__global__ __launch_bounds__(256) void moe_ffn2(
    const __bf16* __restrict__ hbuf, const float* __restrict__ w2,
    const float* __restrict__ b2, const int* __restrict__ lists,
    const int* __restrict__ counts, const int* __restrict__ bases,
    const float* __restrict__ comb, float* __restrict__ out, int N)
{
    const int e     = blockIdx.z;
    const int cnt   = counts[e];
    const int slot0 = blockIdx.y * TM;
    if (slot0 >= cnt) return;
    const int nblk  = blockIdx.x * TN;
    const int base  = bases[e];

    __shared__ __bf16 ldsA[TM][KP];
    __shared__ __bf16 ldsB[TN][KP];
    __shared__ int    ldsTok[TM];
    __shared__ float  ldsCmb[TM];

    const int t    = threadIdx.x;
    const int lane = t & 31;
    const int wid  = t >> 5;

    if (t < TM) {
        int s = slot0 + t;
        if (s < cnt) {
            int tok = lists[(size_t)e * N + s];
            ldsTok[t] = tok;
            ldsCmb[t] = comb[(size_t)tok * MOE_E + e];
        } else {
            ldsTok[t] = 0;
            ldsCmb[t] = 0.0f;
        }
    }
    __syncthreads();

    // A fill: contiguous slots of hbuf (bf16), async copy
    const int arow  = t >> 1;
    const int akoff = (t & 1) * 16;
    int srow = slot0 + arow;
    if (srow >= cnt) srow = cnt - 1;
    const unsigned agb0 = (unsigned)(((size_t)(base + srow) * MOE_H + akoff) * sizeof(__bf16));
    const unsigned alds0 = ldsoff(&ldsA[arow][akoff]);

    const int bn = (t & 15) * 8;
    const int bk = (t >> 4) * 2;
    const float* w2e = w2 + (size_t)e * MOE_H * MOE_D + (size_t)bk * MOE_D + nblk + bn;

    const int wm  = (wid & 3) * 32;
    const int wn  = (wid >> 2) * 64;
    const int l15 = lane & 15;
    const int lh  = lane >> 4;

    v8f acc[2][4] = {};

    for (int k0 = 0; k0 < MOE_H; k0 += KS) {
        const float* wp = w2e + (size_t)k0 * MOE_D;
        float4 r0a = *(const float4*)(wp);
        float4 r0b = *(const float4*)(wp + 4);
        float4 r1a = *(const float4*)(wp + MOE_D);
        float4 r1b = *(const float4*)(wp + MOE_D + 4);

        __syncthreads();

        async_g2l_b128(alds0,      agb0 + (unsigned)(k0 * 2),      hbuf);
        async_g2l_b128(alds0 + 16, agb0 + (unsigned)(k0 * 2) + 16, hbuf);

        float f0[8] = {r0a.x, r0a.y, r0a.z, r0a.w, r0b.x, r0b.y, r0b.z, r0b.w};
        float f1[8] = {r1a.x, r1a.y, r1a.z, r1a.w, r1b.x, r1b.y, r1b.z, r1b.w};
#pragma unroll
        for (int j = 0; j < 8; ++j)
            *(unsigned*)&ldsB[bn + j][bk] = pack_bf16x2(f0[j], f1[j]);

        wait_async0();
        __syncthreads();

        v16bf af[2], bfr[4];
#pragma unroll
        for (int ti = 0; ti < 2; ++ti) {
            int row = wm + ti * 16 + l15;
            v8bf lo = *(const v8bf*)&ldsA[row][lh * 8];
            v8bf hi = *(const v8bf*)&ldsA[row][lh * 8 + 16];
            af[ti] = __builtin_shufflevector(lo, hi, 0,1,2,3,4,5,6,7,8,9,10,11,12,13,14,15);
        }
#pragma unroll
        for (int tj = 0; tj < 4; ++tj) {
            int col = wn + tj * 16 + l15;
            v8bf lo = *(const v8bf*)&ldsB[col][lh * 8];
            v8bf hi = *(const v8bf*)&ldsB[col][lh * 8 + 16];
            bfr[tj] = __builtin_shufflevector(lo, hi, 0,1,2,3,4,5,6,7,8,9,10,11,12,13,14,15);
        }
#pragma unroll
        for (int ti = 0; ti < 2; ++ti)
#pragma unroll
            for (int tj = 0; tj < 4; ++tj)
                acc[ti][tj] = __builtin_amdgcn_wmma_f32_16x16x32_bf16(
                    false, af[ti], false, bfr[tj], (short)0, acc[ti][tj], false, false);
    }

    // epilogue: bias + combine weight, atomic accumulate into out
#pragma unroll
    for (int tj = 0; tj < 4; ++tj) {
        int colg = nblk + wn + tj * 16 + l15;
        float bb = b2[(size_t)e * MOE_D + colg];
#pragma unroll
        for (int ti = 0; ti < 2; ++ti) {
#pragma unroll
            for (int r = 0; r < 8; ++r) {
                int rl = wm + ti * 16 + lh * 8 + r;
                int s  = slot0 + rl;
                if (s < cnt) {
                    float v = acc[ti][tj][r] + bb;
                    atomicAdd(&out[(size_t)ldsTok[rl] * MOE_D + colg], ldsCmb[rl] * v);
                }
            }
        }
    }
}

// =====================================================================
// launch
// =====================================================================
extern "C" void kernel_launch(void* const* d_in, const int* in_sizes, int n_in,
                              void* d_out, int out_size, void* d_ws, size_t ws_size,
                              hipStream_t stream)
{
    (void)n_in; (void)out_size; (void)ws_size;
    const float* x  = (const float*)d_in[0];
    const float* gw = (const float*)d_in[1];
    const float* gb = (const float*)d_in[2];
    const float* w1 = (const float*)d_in[3];
    const float* b1 = (const float*)d_in[4];
    const float* w2 = (const float*)d_in[5];
    const float* b2 = (const float*)d_in[6];
    float* out = (float*)d_out;

    const int N = in_sizes[0] / MOE_D;   // B*T tokens

    // workspace layout
    char* ws = (char*)d_ws;
    int*   counts = (int*)ws;                       // [8]
    int*   bases  = counts + MOE_E;                 // [8]
    size_t off = 256;
    float* comb  = (float*)(ws + off);  off += (size_t)N * MOE_E * sizeof(float);
    int*   lists = (int*)(ws + off);    off += (size_t)N * MOE_E * sizeof(int);
    __bf16* xb   = (__bf16*)(ws + off); off += (size_t)N * MOE_D * sizeof(__bf16);
    __bf16* hbuf = (__bf16*)(ws + off); // [N*K][H] bf16

    const size_t total = (size_t)N * MOE_D;
    moe_prep<<<dim3((unsigned)((total + 255) / 256)), 256, 0, stream>>>(x, xb, out, counts, total);
    moe_router<<<dim3((unsigned)N), 32, 0, stream>>>(x, gw, gb, comb, lists, counts, N);
    moe_bases<<<1, 1, 0, stream>>>(counts, bases);

    dim3 g1(MOE_H / TN, (unsigned)((N + TM - 1) / TM), MOE_E);
    moe_ffn1<<<g1, 256, 0, stream>>>(xb, w1, b1, lists, counts, bases, hbuf, N);

    dim3 g2(MOE_D / TN, (unsigned)((N + TM - 1) / TM), MOE_E);
    moe_ffn2<<<g2, 256, 0, stream>>>(hbuf, w2, b2, lists, counts, bases, comb, out, N);
}
